// Attention_6708738916658
// MI455X (gfx1250) — matrix operand order repaired and checked by reading, not compiled, not run on hardware
//
#include <hip/hip_runtime.h>
#include <hip/hip_bf16.h>
#include <math.h>

typedef __attribute__((ext_vector_type(16))) _Float16 v16h;
typedef __attribute__((ext_vector_type(8)))  float    v8f;

#define NHEAD 12
#define HD    64
#define NTOK  1025
#define NPAD  1088            // 17*64: key loop steps by 64
#define BATCH 8
#define CDIM  768
#define C3    2304
#define MROWS (BATCH*NTOK)    // 8200
#define MPAD  8256            // 129*64: GEMM M tiles of 64
#define ATT_SCALE 0.125f      // 64^-0.5

// ---------------- WMMA fragment helpers (contiguous k order, k = 16*h + s, A and B alike) ----------

// A-matrix fragment: 16(M) x 32(K), element (m,k) at p[m*ld + k] (f16 row-major).
// Lanes 0-15 / 16-31 both hold M=0..15; low half holds K 0..15, high
// half K 16..31, two consecutive K per VGPR -> two b128 loads per lane.
__device__ __forceinline__ v16h load_a_frag(const _Float16* p, int ld) {
  const int lane = threadIdx.x & 31;
  const int m  = lane & 15;
  const int hi = lane >> 4;
  v16h a;
#pragma unroll
  for (int i = 0; i < 8; ++i) {
    int k = 2 * i + hi * 16;
    a[2 * i]     = p[m * ld + k];
    a[2 * i + 1] = p[m * ld + k + 1];
  }
  return a;
}

// B-matrix fragment where memory holds B transposed: B[k][n] = p[n*ld + k]
// (row-major weight W[n][k] for x@W^T, or row-major K[token][d] for Q@K^T).
__device__ __forceinline__ v16h load_bT_frag(const _Float16* p, int ld) {
  const int lane = threadIdx.x & 31;
  const int n  = lane & 15;
  const int hi = lane >> 4;
  v16h b;
#pragma unroll
  for (int j = 0; j < 8; ++j) {
    int k = 2 * j + hi * 16;
    b[2 * j]     = p[n * ld + k];
    b[2 * j + 1] = p[n * ld + k + 1];
  }
  return b;
}

// B-matrix fragment from row-major [k][n]: B[k][n] = p[k*ld + n]  (V matrix).
__device__ __forceinline__ v16h load_b_frag(const _Float16* p, int ld) {
  const int lane = threadIdx.x & 31;
  const int n  = lane & 15;
  const int hi = lane >> 4;
  v16h b;
#pragma unroll
  for (int j = 0; j < 8; ++j) {
    int k = 2 * j + hi * 16;
    b[2 * j]     = p[k * ld + n];
    b[2 * j + 1] = p[(k + 1) * ld + n];
  }
  return b;
}

__device__ __forceinline__ v8f wmma_f16(v16h a, v16h b, v8f c) {
  return __builtin_amdgcn_wmma_f32_16x16x32_f16(false, a, false, b,
                                                (short)0, c, false, false);
}

// ---------------- Kernel 1: QKV GEMM + bias + RoPE -> f16 q/k/v -------------
// Grid (129, 36), block 128 (4 waves). Block tile 64(M) x 64(N); wave w owns
// M-strip w*16 and all 64 N cols (4 accumulators, A-frag reused 4x).
// Double-buffered LDS tiles: global loads for tile i+1 overlap WMMA on tile i.
__global__ __launch_bounds__(128)
void qkv_rope_kernel(const float* __restrict__ x, const float* __restrict__ sin_t,
                     const float* __restrict__ cos_t, const float* __restrict__ qkv_w,
                     const float* __restrict__ qkv_b,
                     _Float16* __restrict__ q_ws, _Float16* __restrict__ k_ws,
                     _Float16* __restrict__ v_ws) {
  __shared__ _Float16 lA[2][64 * 40];
  __shared__ _Float16 lB[2][64 * 40];
  __shared__ float    lO[64 * 66];

  const int tid  = threadIdx.x;
  const int wave = tid >> 5;
  const int m0 = blockIdx.x * 64;
  const int n0 = blockIdx.y * 64;

  float4 ra[4], rb[4];
  // Issue: global -> regs (8 b128 loads back-to-back, row index clamped so no
  // divergent masking; clamped rows only feed never-written output rows).
  auto issue = [&](int k0) {
#pragma unroll
    for (int u = 0; u < 4; ++u) {
      int i = tid + u * 128;
      int r = i >> 3, c4 = (i & 7) * 4;
      int grc = min(m0 + r, MROWS - 1);
      ra[u] = *(const float4*)(x + (size_t)grc * CDIM + k0 + c4);
      rb[u] = *(const float4*)(qkv_w + (size_t)(n0 + r) * CDIM + k0 + c4);
    }
  };
  // Commit: regs -> f16 LDS tiles.
  auto commit = [&](int cur) {
#pragma unroll
    for (int u = 0; u < 4; ++u) {
      int i = tid + u * 128;
      int r = i >> 3, c4 = (i & 7) * 4;
      _Float16* pa = &lA[cur][r * 40 + c4];
      pa[0] = (_Float16)ra[u].x; pa[1] = (_Float16)ra[u].y;
      pa[2] = (_Float16)ra[u].z; pa[3] = (_Float16)ra[u].w;
      _Float16* pb = &lB[cur][r * 40 + c4];
      pb[0] = (_Float16)rb[u].x; pb[1] = (_Float16)rb[u].y;
      pb[2] = (_Float16)rb[u].z; pb[3] = (_Float16)rb[u].w;
    }
  };

  v8f acc[4] = {{}, {}, {}, {}};
  issue(0);
  for (int k0 = 0, it = 0; k0 < CDIM; k0 += 32, ++it) {
    const int cur = it & 1;
    commit(cur);
    __syncthreads();
    if (k0 + 32 < CDIM) issue(k0 + 32);   // overlap with WMMA below
    v16h a = load_a_frag(&lA[cur][wave * 16 * 40], 40);
#pragma unroll
    for (int c = 0; c < 4; ++c) {
      v16h b = load_bT_frag(&lB[cur][c * 16 * 40], 40);
      acc[c] = wmma_f16(a, b, acc[c]);
    }
  }
  __syncthreads();

  // Spill the 64x64 fp32 tile to LDS so RoPE can read the d^32 partner.
  {
    const int lane  = tid & 31;
    const int colb  = lane & 15;
    const int rbase = wave * 16 + (lane >> 4) * 8;
#pragma unroll
    for (int c = 0; c < 4; ++c)
#pragma unroll
      for (int j = 0; j < 8; ++j)
        lO[(rbase + j) * 66 + c * 16 + colb] = acc[c][j];
  }
  __syncthreads();

  const int which = n0 / CDIM;          // 0=q, 1=k, 2=v
  const int head  = (n0 % CDIM) / HD;
  _Float16* dst = (which == 0) ? q_ws : (which == 1) ? k_ws : v_ws;

  for (int i = tid; i < 64 * 64; i += 128) {
    int r = i >> 6, d = i & 63;
    int gr = m0 + r;
    if (gr >= MROWS) continue;
    int bb  = gr / NTOK;
    int tok = gr % NTOK;
    float val = lO[r * 66 + d] + qkv_b[n0 + d];
    if (which < 2 && tok > 0) {           // RoPE on patch tokens of q,k
      float sn = sin_t[(tok - 1) * HD + d];
      float cs = cos_t[(tok - 1) * HD + d];
      float part = lO[r * 66 + (d ^ 32)] + qkv_b[n0 + (d ^ 32)];
      float rot  = (d < 32) ? -part : part;
      val = val * cs + rot * sn;
    }
    dst[((size_t)(bb * NHEAD + head) * NPAD + tok) * HD + d] = (_Float16)val;
  }
}

// ---------------- Kernel 2: flash attention, 1 wave per 16-row Q tile -------
// Grid (65, 96), block 32. Key step 64: 8 WMMA (QK^T) + 8 WMMA (PV) per step.
__global__ __launch_bounds__(32)
void attn_kernel(const _Float16* __restrict__ q_ws, const _Float16* __restrict__ k_ws,
                 const _Float16* __restrict__ v_ws, _Float16* __restrict__ ao_ws) {
  __shared__ _Float16 lP[16 * 72];      // P tile 16x64, stride 72

  const int lane = threadIdx.x & 31;
  const int qt = blockIdx.x;            // 0..64 (tile 64 partially valid)
  const int bh = blockIdx.y;            // b*12 + h
  const _Float16* qb = q_ws + (size_t)bh * NPAD * HD + (size_t)qt * 16 * HD;
  const _Float16* kb = k_ws + (size_t)bh * NPAD * HD;
  const _Float16* vb = v_ws + (size_t)bh * NPAD * HD;

  const v16h qa0 = load_a_frag(qb, HD);       // d = 0..31
  const v16h qa1 = load_a_frag(qb + 32, HD);  // d = 32..63

  v8f acc[4] = {{}, {}, {}, {}};
  float mrow[8], lrow[8];
#pragma unroll
  for (int j = 0; j < 8; ++j) { mrow[j] = -1e30f; lrow[j] = 0.f; }

  const int col = lane & 15;
  const int rbase = (lane >> 4) * 8;

  for (int kt = 0; kt < NPAD; kt += 64) {
    __builtin_prefetch(kb + (size_t)(kt + 64) * HD, 0, 0);
    // S = Q @ K^T : four 16x16 key sub-tiles, K-dim 64 in two chunks each
    float p[4][8];
#pragma unroll
    for (int t = 0; t < 4; ++t) {
      const _Float16* kbase = kb + (size_t)(kt + t * 16) * HD;
      v8f s = {};
      s = wmma_f16(qa0, load_bT_frag(kbase,      HD), s);
      s = wmma_f16(qa1, load_bT_frag(kbase + 32, HD), s);
      const bool ok = (kt + t * 16 + col) < NTOK;
#pragma unroll
      for (int j = 0; j < 8; ++j)
        p[t][j] = ok ? s[j] * ATT_SCALE : -1e30f;
    }
    // Online softmax: rows live in 16-lane halves -> xor-reduce over 1,2,4,8.
#pragma unroll
    for (int j = 0; j < 8; ++j) {
      float t = fmaxf(fmaxf(p[0][j], p[1][j]), fmaxf(p[2][j], p[3][j]));
      t = fmaxf(t, __shfl_xor(t, 1));
      t = fmaxf(t, __shfl_xor(t, 2));
      t = fmaxf(t, __shfl_xor(t, 4));
      t = fmaxf(t, __shfl_xor(t, 8));
      float mnew = fmaxf(mrow[j], t);
      float corr = __expf(mrow[j] - mnew);
      p[0][j] = __expf(p[0][j] - mnew);
      p[1][j] = __expf(p[1][j] - mnew);
      p[2][j] = __expf(p[2][j] - mnew);
      p[3][j] = __expf(p[3][j] - mnew);
      float rs = (p[0][j] + p[1][j]) + (p[2][j] + p[3][j]);
      rs += __shfl_xor(rs, 1);
      rs += __shfl_xor(rs, 2);
      rs += __shfl_xor(rs, 4);
      rs += __shfl_xor(rs, 8);
      lrow[j] = lrow[j] * corr + rs;
      mrow[j] = mnew;
      acc[0][j] *= corr; acc[1][j] *= corr; acc[2][j] *= corr; acc[3][j] *= corr;
    }
    // P (C-layout) -> LDS -> A-layout for the PV matmul.
#pragma unroll
    for (int t = 0; t < 4; ++t)
#pragma unroll
      for (int j = 0; j < 8; ++j)
        lP[(rbase + j) * 72 + t * 16 + col] = (_Float16)p[t][j];
    __syncthreads();
    v16h pa0 = load_a_frag(lP, 72);       // keys kt .. kt+31
    v16h pa1 = load_a_frag(lP + 32, 72);  // keys kt+32 .. kt+63
    const _Float16* v0 = vb + (size_t)kt * HD;
    const _Float16* v1 = vb + (size_t)(kt + 32) * HD;
#pragma unroll
    for (int c = 0; c < 4; ++c) {
      acc[c] = wmma_f16(pa0, load_b_frag(v0 + c * 16, HD), acc[c]);
      acc[c] = wmma_f16(pa1, load_b_frag(v1 + c * 16, HD), acc[c]);
    }
    __syncthreads();
  }

  // Epilogue: normalize, write [row = b*1025+tok][h*64 + d] f16 buffer.
  const int bb = bh / NHEAD, head = bh % NHEAD;
#pragma unroll
  for (int j = 0; j < 8; ++j) {
    int tok = qt * 16 + rbase + j;
    if (tok >= NTOK) continue;
    float inv = 1.f / lrow[j];
    size_t base = (size_t)(bb * NTOK + tok) * CDIM + head * HD;
#pragma unroll
    for (int c = 0; c < 4; ++c)
      ao_ws[base + c * 16 + col] = (_Float16)(acc[c][j] * inv);
  }
}

// ---------------- Kernel 3: output projection (fp32 out) --------------------
// Grid (129, 12), block 128. 64x64 block tile, double-buffered staging.
__global__ __launch_bounds__(128)
void proj_kernel(const _Float16* __restrict__ ao_ws, const float* __restrict__ proj_w,
                 const float* __restrict__ proj_b, float* __restrict__ out) {
  __shared__ _Float16 lA[2][64 * 40];
  __shared__ _Float16 lB[2][64 * 40];

  const int tid  = threadIdx.x;
  const int wave = tid >> 5;
  const int m0 = blockIdx.x * 64;
  const int n0 = blockIdx.y * 64;

  uint4  raw[2];     // A tile: 2 x 8-half vectors per thread (f16 copy)
  float4 rb[4];      // B tile: 4 x float4 per thread (fp32 -> f16)
  auto issue = [&](int k0) {
#pragma unroll
    for (int u = 0; u < 2; ++u) {
      int i = tid + u * 128;
      int r = i >> 2, c8 = (i & 3) * 8;
      raw[u] = *(const uint4*)(ao_ws + (size_t)(m0 + r) * CDIM + k0 + c8);
    }
#pragma unroll
    for (int u = 0; u < 4; ++u) {
      int i = tid + u * 128;
      int r = i >> 3, c4 = (i & 7) * 4;
      rb[u] = *(const float4*)(proj_w + (size_t)(n0 + r) * CDIM + k0 + c4);
    }
  };
  auto commit = [&](int cur) {
#pragma unroll
    for (int u = 0; u < 2; ++u) {
      int i = tid + u * 128;
      int r = i >> 2, c8 = (i & 3) * 8;
      *(uint4*)(&lA[cur][r * 40 + c8]) = raw[u];
    }
#pragma unroll
    for (int u = 0; u < 4; ++u) {
      int i = tid + u * 128;
      int r = i >> 3, c4 = (i & 7) * 4;
      _Float16* pb = &lB[cur][r * 40 + c4];
      pb[0] = (_Float16)rb[u].x; pb[1] = (_Float16)rb[u].y;
      pb[2] = (_Float16)rb[u].z; pb[3] = (_Float16)rb[u].w;
    }
  };

  v8f acc[4] = {{}, {}, {}, {}};
  issue(0);
  for (int k0 = 0, it = 0; k0 < CDIM; k0 += 32, ++it) {
    const int cur = it & 1;
    commit(cur);
    __syncthreads();
    if (k0 + 32 < CDIM) issue(k0 + 32);
    v16h a = load_a_frag(&lA[cur][wave * 16 * 40], 40);
#pragma unroll
    for (int c = 0; c < 4; ++c) {
      v16h b = load_bT_frag(&lB[cur][c * 16 * 40], 40);
      acc[c] = wmma_f16(a, b, acc[c]);
    }
  }

  // Direct register -> global epilogue with bias.
  const int lane  = tid & 31;
  const int colb  = lane & 15;
  const int rbase = (lane >> 4) * 8;
  const int grow0 = m0 + wave * 16 + rbase;
#pragma unroll
  for (int c = 0; c < 4; ++c) {
    const int colg = n0 + c * 16 + colb;
    const float bias = proj_b[colg];
#pragma unroll
    for (int j = 0; j < 8; ++j) {
      int gr = grow0 + j;
      if (gr < MROWS) out[(size_t)gr * CDIM + colg] = acc[c][j] + bias;
    }
  }
}

// ---------------- Launch ----------------------------------------------------
extern "C" void kernel_launch(void* const* d_in, const int* in_sizes, int n_in,
                              void* d_out, int out_size, void* d_ws, size_t ws_size,
                              hipStream_t stream) {
  (void)in_sizes; (void)n_in; (void)out_size; (void)ws_size;
  const float* x      = (const float*)d_in[0];
  const float* sin_t  = (const float*)d_in[1];
  const float* cos_t  = (const float*)d_in[2];
  const float* qkv_w  = (const float*)d_in[3];
  const float* qkv_b  = (const float*)d_in[4];
  const float* proj_w = (const float*)d_in[5];
  const float* proj_b = (const float*)d_in[6];
  float* out = (float*)d_out;

  const size_t qkv_elems = (size_t)BATCH * NHEAD * NPAD * HD;
  _Float16* q_ws  = (_Float16*)d_ws;
  _Float16* k_ws  = q_ws + qkv_elems;
  _Float16* v_ws  = k_ws + qkv_elems;
  _Float16* ao_ws = v_ws + qkv_elems;                  // MPAD x 768 f16

  qkv_rope_kernel<<<dim3(MPAD / 64, C3 / 64), dim3(128), 0, stream>>>(
      x, sin_t, cos_t, qkv_w, qkv_b, q_ws, k_ws, v_ws);
  attn_kernel<<<dim3(65, BATCH * NHEAD), dim3(32), 0, stream>>>(
      q_ws, k_ws, v_ws, ao_ws);
  proj_kernel<<<dim3(MPAD / 64, CDIM / 64), dim3(128), 0, stream>>>(
      ao_ws, proj_w, proj_b, out);
}
